// HeadsSelection_15590731285029
// MI455X (gfx1250) — matrix operand order repaired and checked by reading, not compiled, not run on hardware
//
#include <hip/hip_runtime.h>
#include <hip/hip_bf16.h>

// ---------- WMMA types (CDNA5 / gfx1250, wave32) ----------
typedef __attribute__((ext_vector_type(16))) __bf16 v16bf;
typedef __attribute__((ext_vector_type(8)))  float  v8f;

union FragU { uint4 u4[2]; v16bf v; };

// float -> bf16 (RNE)
__device__ __forceinline__ unsigned short f2bf(float f) {
    unsigned int u = __float_as_uint(f);
    unsigned int r = u + 0x7fffu + ((u >> 16) & 1u);
    return (unsigned short)(r >> 16);
}

__device__ __forceinline__ float sigf(float x) { return 1.0f / (1.0f + __expf(-x)); }
__device__ __forceinline__ float tanh_fast(float x) { return 2.0f * sigf(2.0f * x) - 1.0f; }

// A fragment: 16x32 bf16, A row-major (lda elems). ISA layout:
// lanes 0-15: M=lane, elems = K {k0..k0+7, k0+16..k0+23}
// lanes 16-31: M=lane-16, elems = K {k0+8..k0+15, k0+24..k0+31}
__device__ __forceinline__ v16bf load_a_frag(const unsigned short* A, int lda,
                                             int m0, int k0, int lane) {
    int row = m0 + (lane & 15);
    int kb  = k0 + ((lane & 16) ? 8 : 0);
    const unsigned short* p = A + (size_t)row * lda + kb;
    FragU f;
    f.u4[0] = *(const uint4*)(p);
    f.u4[1] = *(const uint4*)(p + 16);
    return f.v;
}

// B fragment: 32x16 (KxN) bf16, from weight stored row-major (N x K, ldw elems).
// lanes 0-15: col=lane, K={k0..k0+7, k0+16..k0+23} ; lanes 16-31: col=lane-16, K={k0+8..k0+15, k0+24..k0+31}
__device__ __forceinline__ v16bf load_b_frag(const unsigned short* W, int ldw,
                                             int n0, int k0, int lane) {
    int row = n0 + (lane & 15);
    int kb  = k0 + ((lane & 16) ? 8 : 0);
    const unsigned short* p = W + (size_t)row * ldw + kb;
    FragU f;
    f.u4[0] = *(const uint4*)(p);
    f.u4[1] = *(const uint4*)(p + 16);
    return f.v;
}

// ---------- Kernel 0: convert weights to bf16, pack [Ua;pad;Wa;pad] ----------
__global__ __launch_bounds__(256) void pack_weights_kernel(
    const float* Wih_f, const float* Whh_f, const float* Wih_b, const float* Whh_b,
    const float* Ua, const float* Wa,
    unsigned short* wihf, unsigned short* whhf, unsigned short* wihb,
    unsigned short* whhb, unsigned short* uawa)
{
    int idx = blockIdx.x * 256 + threadIdx.x;
    int which = blockIdx.y;
    if (which < 4) {
        if (idx < 1024 * 256) {
            const float* s = (which == 0) ? Wih_f : (which == 1) ? Whh_f
                           : (which == 2) ? Wih_b : Whh_b;
            unsigned short* d = (which == 0) ? wihf : (which == 1) ? whhf
                              : (which == 2) ? wihb : whhb;
            d[idx] = f2bf(s[idx]);
        }
    } else {
        // uawa: 64 rows x 512 cols. rows 0..29 = Ua, 32..61 = Wa, rest 0
        if (idx < 64 * 512) {
            int n = idx >> 9, k = idx & 511;
            float v = 0.0f;
            if (n < 30)                 v = Ua[n * 512 + k];
            else if (n >= 32 && n < 62) v = Wa[(n - 32) * 512 + k];
            uawa[idx] = f2bf(v);
        }
    }
}

// ---------- Kernel 1: embedding gather (fwd + reversed-seq for bwd) ----------
__global__ __launch_bounds__(256) void gather_embed_kernel(
    const int* concepts, const int* lens, const float* emb,
    unsigned short* ebf_f, unsigned short* ebf_b)
{
    int r = blockIdx.x;            // r = t*64 + b
    int t = r >> 6, b = r & 63;
    int e = threadIdx.x;           // 0..255
    int len = lens[b];
    int rt  = (t < len) ? (len - 1 - t) : t;
    int tok_f = concepts[t * 64 + b];
    int tok_b = concepts[rt * 64 + b];
    ebf_f[(size_t)r * 256 + e] = f2bf(emb[(size_t)tok_f * 256 + e]);
    ebf_b[(size_t)r * 256 + e] = f2bf(emb[(size_t)tok_b * 256 + e]);
}

// ---------- Kernel 2: generic WMMA GEMM  C[MxN] = A[MxK](bf16) * W^T + bias ----------
// W stored row-major (N x K) bf16. Block = 4 waves, 64x64 tile per block.
__global__ __launch_bounds__(128) void wmma_gemm_bf16_kernel(
    const unsigned short* A, const unsigned short* W, const float* bias,
    float* C, int M, int N, int K)
{
    int lane = threadIdx.x & 31;
    int wave = threadIdx.x >> 5;               // 0..3 -> M sub-tile
    int m0 = blockIdx.x * 64 + wave * 16;
    int n0 = blockIdx.y * 64;
    v8f acc[4] = {};
    for (int k0 = 0; k0 < K; k0 += 32) {
        v16bf a = load_a_frag(A, K, m0, k0, lane);
#pragma unroll
        for (int nt = 0; nt < 4; nt++) {
            v16bf bfrag = load_b_frag(W, K, n0 + nt * 16, k0, lane);
            acc[nt] = __builtin_amdgcn_wmma_f32_16x16x32_bf16(
                false, a, false, bfrag, (short)0, acc[nt], false, false);
        }
    }
    int col_in = lane & 15;
    int rbase  = m0 + ((lane & 16) ? 8 : 0);
#pragma unroll
    for (int nt = 0; nt < 4; nt++) {
        int col = n0 + nt * 16 + col_in;
        float bv = bias ? bias[col] : 0.0f;
#pragma unroll
        for (int e = 0; e < 8; e++)
            C[(size_t)(rbase + e) * N + col] = acc[nt][e] + bv;
    }
}

// ---------- Kernel 3: LSTM recurrence (one 32-wave block per direction) ----------
// grid.x = 2 (dir). h lives in LDS (bf16, 64x256 = 32KB), c lives in VGPRs.
// Each wave: batch tile mt (16 rows), hidden cols [32*hq, 32*hq+32) -> computes
// its 4 gate tiles per hidden sub-tile so the cell update is wave-local.
__global__ __launch_bounds__(1024) void lstm_wmma_kernel(
    const float* xg_f, const float* xg_b,
    const unsigned short* whh_f, const unsigned short* whh_b,
    float* hs_f, float* hs_b)
{
    __shared__ unsigned short hsh[64 * 256];   // h_t in bf16

    const float* xg          = (blockIdx.x == 0) ? xg_f  : xg_b;
    const unsigned short* Wh = (blockIdx.x == 0) ? whh_f : whh_b;
    float* hs                = (blockIdx.x == 0) ? hs_f  : hs_b;

    int lane = threadIdx.x & 31;
    int wave = threadIdx.x >> 5;   // 0..31
    int mt = wave & 3;             // batch tile
    int hq = wave >> 2;            // hidden group: cols [32*hq, 32*hq+32)

    // zero h in LDS
    unsigned int* z = (unsigned int*)hsh;
    for (int i = threadIdx.x; i < 64 * 256 / 2; i += 1024) z[i] = 0u;
    __syncthreads();

    v8f creg[2] = {};              // cell state, persistent across all 128 steps

    for (int t = 0; t < 128; t++) {
        if (t + 1 < 128)           // warm L2/WGP$ for next step's gate inputs
            __builtin_prefetch(xg + (size_t)(t + 1) * 64 * 1024 + threadIdx.x * 64, 0, 1);

        v8f acc[2][4] = {};
        for (int k0 = 0; k0 < 256; k0 += 32) {
            // A fragment: h tile (16x32) from LDS
            int row = mt * 16 + (lane & 15);
            int kb  = k0 + ((lane & 16) ? 8 : 0);
            const unsigned short* p = hsh + row * 256 + kb;
            FragU fa;
            fa.u4[0] = *(const uint4*)(p);
            fa.u4[1] = *(const uint4*)(p + 16);
#pragma unroll
            for (int qt = 0; qt < 2; qt++) {
#pragma unroll
                for (int g = 0; g < 4; g++) {
                    int ntile = g * 16 + hq * 2 + qt;   // gate g, hidden tile
                    v16bf bfrag = load_b_frag(Wh, 256, ntile * 16, k0, lane);
                    acc[qt][g] = __builtin_amdgcn_wmma_f32_16x16x32_bf16(
                        false, fa.v, false, bfrag, (short)0, acc[qt][g], false, false);
                }
            }
        }
        __syncthreads();   // all waves done reading h_t before anyone overwrites

        int coll  = lane & 15;
        int rbase = mt * 16 + ((lane & 16) ? 8 : 0);
#pragma unroll
        for (int qt = 0; qt < 2; qt++) {
            int hcol = hq * 32 + qt * 16 + coll;
#pragma unroll
            for (int e = 0; e < 8; e++) {
                int m = rbase + e;
                size_t xrow = (size_t)(t * 64 + m) * 1024;
                float gi = acc[qt][0][e] + xg[xrow + 0 * 256 + hcol];
                float gf = acc[qt][1][e] + xg[xrow + 1 * 256 + hcol];
                float gg = acc[qt][2][e] + xg[xrow + 2 * 256 + hcol];
                float go = acc[qt][3][e] + xg[xrow + 3 * 256 + hcol];
                float c  = sigf(gf) * creg[qt][e] + sigf(gi) * tanh_fast(gg);
                creg[qt][e] = c;
                float h = sigf(go) * tanh_fast(c);
                hsh[m * 256 + hcol] = f2bf(h);
                hs[(size_t)(t * 64 + m) * 256 + hcol] = h;
            }
        }
        __syncthreads();   // h_{t+1} visible before next step's A loads
    }
}

// ---------- Kernel 4: build enc (bf16), un-reverse bwd, apply valid mask ----------
__global__ __launch_bounds__(256) void combine_enc_kernel(
    const float* hs_f, const float* hs_b, const int* lens, unsigned short* enc)
{
    int r = blockIdx.x;            // r = b*128 + t  (matches edge-score indexing)
    int b = r >> 7, t = r & 127;
    int k = threadIdx.x;           // 0..255
    int len = lens[b];
    bool valid = (t < len);
    int rev = valid ? (len - 1 - t) : t;
    float fv = valid ? hs_f[((size_t)(t   * 64 + b)) * 256 + k] : 0.0f;
    float bv = valid ? hs_b[((size_t)(rev * 64 + b)) * 256 + k] : 0.0f;
    enc[(size_t)r * 512 + k]       = f2bf(fv);
    enc[(size_t)r * 512 + 256 + k] = f2bf(bv);
}

// ---------- Kernel 5: pairwise edge scores ----------
// uw: (8192 x 64) f32; cols 0..29 = u, 32..61 = w.
__global__ __launch_bounds__(128) void edge_scores_kernel(
    const float* uw, const float* va, float* scores, float* preds)
{
    __shared__ float us[32];
    int bi = blockIdx.x;           // b*128 + i
    if (threadIdx.x < 30) us[threadIdx.x] = uw[(size_t)bi * 64 + threadIdx.x];
    __syncthreads();
    int b = bi >> 7;
    int j = threadIdx.x;
    const float* wrow = uw + (size_t)(b * 128 + j) * 64 + 32;
    float s = 0.0f;
#pragma unroll
    for (int k = 0; k < 30; k++)
        s += va[k] * tanh_fast(us[k] + wrow[k]);
    size_t idx = (size_t)bi * 128 + j;
    scores[idx] = s;
    preds[idx]  = (s >= 0.0f) ? 1.0f : 0.0f;   // sigmoid(s)>=0.5 <=> s>=0
}

// ---------- Host launch ----------
static inline size_t alignup(size_t x) { return (x + 255) & ~(size_t)255; }

extern "C" void kernel_launch(void* const* d_in, const int* in_sizes, int n_in,
                              void* d_out, int out_size, void* d_ws, size_t ws_size,
                              hipStream_t stream) {
    (void)in_sizes; (void)n_in; (void)out_size; (void)ws_size;
    const int*   concepts = (const int*)  d_in[0];
    const int*   lens     = (const int*)  d_in[1];
    const float* emb      = (const float*)d_in[2];
    const float* Wih_f    = (const float*)d_in[3];
    const float* Whh_f    = (const float*)d_in[4];
    const float* b_f      = (const float*)d_in[5];
    const float* Wih_b    = (const float*)d_in[6];
    const float* Whh_b    = (const float*)d_in[7];
    const float* b_b      = (const float*)d_in[8];
    const float* Ua       = (const float*)d_in[9];
    const float* Wa       = (const float*)d_in[10];
    const float* va       = (const float*)d_in[11];

    // Workspace carve-up
    char* ws = (char*)d_ws;
    size_t off = 0;
    auto take = [&](size_t bytes) { char* p = ws + off; off += alignup(bytes); return p; };
    unsigned short* wihf  = (unsigned short*)take(1024 * 256 * 2);
    unsigned short* whhf  = (unsigned short*)take(1024 * 256 * 2);
    unsigned short* wihb  = (unsigned short*)take(1024 * 256 * 2);
    unsigned short* whhb  = (unsigned short*)take(1024 * 256 * 2);
    unsigned short* uawa  = (unsigned short*)take(64 * 512 * 2);
    unsigned short* ebf_f = (unsigned short*)take((size_t)8192 * 256 * 2);
    unsigned short* ebf_b = (unsigned short*)take((size_t)8192 * 256 * 2);
    float*          xg_f  = (float*)take((size_t)8192 * 1024 * 4);
    float*          xg_b  = (float*)take((size_t)8192 * 1024 * 4);
    float*          hs_f  = (float*)take((size_t)8192 * 256 * 4);
    float*          hs_b  = (float*)take((size_t)8192 * 256 * 4);
    unsigned short* encb  = (unsigned short*)take((size_t)8192 * 512 * 2);
    float*          uw    = (float*)take((size_t)8192 * 64 * 4);

    float* scores = (float*)d_out;
    float* preds  = scores + (size_t)64 * 128 * 128;

    // 0) weights -> bf16 (+ packed [Ua;Wa])
    pack_weights_kernel<<<dim3(1024, 5), 256, 0, stream>>>(
        Wih_f, Whh_f, Wih_b, Whh_b, Ua, Wa, wihf, whhf, wihb, whhb, uawa);

    // 1) embedding gather (fwd + reversed)
    gather_embed_kernel<<<8192, 256, 0, stream>>>(concepts, lens, emb, ebf_f, ebf_b);

    // 2) input projections: xg = emb @ Wih^T + b   (M=8192, N=1024, K=256)
    wmma_gemm_bf16_kernel<<<dim3(128, 16), 128, 0, stream>>>(
        ebf_f, wihf, b_f, xg_f, 8192, 1024, 256);
    wmma_gemm_bf16_kernel<<<dim3(128, 16), 128, 0, stream>>>(
        ebf_b, wihb, b_b, xg_b, 8192, 1024, 256);

    // 3) both LSTM recurrences (one resident 32-wave block each)
    lstm_wmma_kernel<<<2, 1024, 0, stream>>>(xg_f, xg_b, whhf, whhb, hs_f, hs_b);

    // 4) enc = concat(fwd, unrev(bwd)) * valid, bf16
    combine_enc_kernel<<<8192, 256, 0, stream>>>(hs_f, hs_b, lens, encb);

    // 5) u/w projections: uw = enc @ [Ua;Wa]^T  (M=8192, N=64, K=512)
    wmma_gemm_bf16_kernel<<<dim3(128, 1), 128, 0, stream>>>(
        encb, uawa, nullptr, uw, 8192, 64, 512);

    // 6) pairwise scores + predictions
    edge_scores_kernel<<<8192, 128, 0, stream>>>(uw, va, scores, preds);
}
